// PointROPEAttention_39092792328858
// MI455X (gfx1250) — matrix operand order repaired and checked by reading, not compiled, not run on hardware
//
#include <hip/hip_runtime.h>
#include <hip/hip_bf16.h>

// ---------------- problem constants ----------------
#define N_PTS 65536
#define CCH   768
#define NH    8
#define DH    96        // head dim
#define PK    256       // patch size
#define NPATCH (N_PTS / PK)
#define SCALE_ATTN 0.1020620726159658f   // 96^-0.5

// ---------------- types ----------------
typedef __bf16 bf16_t;
typedef __attribute__((ext_vector_type(16))) __bf16 v16bf;
typedef __attribute__((ext_vector_type(8)))  float  v8f;
typedef __attribute__((ext_vector_type(4)))  unsigned int u32x4;
typedef __attribute__((ext_vector_type(4)))  int i32x4;

union V16 {
    v16bf v;
    u32x4 q[2];
    unsigned short s[16];
    bf16_t b[16];
};

__device__ __forceinline__ bf16_t f2bf(float f) {
    union { float f; unsigned u; } x; x.f = f;
    unsigned r = x.u + 0x7FFFu + ((x.u >> 16) & 1u);   // RNE
    union { unsigned short s; bf16_t b; } y;
    y.s = (unsigned short)(r >> 16);
    return y.b;
}

// ---------------- CDNA5 async global->LDS copy (ASYNCcnt path) ----------------
#if __has_builtin(__builtin_amdgcn_global_load_async_to_lds_b128)
#define ASYNC_MODE 2
#pragma message("CDNA5: using __builtin_amdgcn_global_load_async_to_lds_b128")
#elif __has_builtin(__builtin_amdgcn_global_load_lds)
#define ASYNC_MODE 1
#pragma message("CDNA5: async builtin missing; using __builtin_amdgcn_global_load_lds")
#else
#define ASYNC_MODE 0
#pragma message("CDNA5: no global->LDS builtins; manual staging")
#endif

// copy 16 bytes global -> LDS without touching VGPRs (when supported)
// builtin signature (from hipcc diagnostic): arg0 = AS1 (global) int4*,
// then LDS pointer, imm offset, imm cpol.
__device__ __forceinline__ void cp_async16(void* lds, const void* g) {
#if ASYNC_MODE == 2
    __builtin_amdgcn_global_load_async_to_lds_b128(
        (__attribute__((address_space(1))) i32x4*)(unsigned long long)(uintptr_t)g,
        (__attribute__((address_space(3))) i32x4*)(unsigned)(uintptr_t)lds,
        0, 0);
#elif ASYNC_MODE == 1
    __builtin_amdgcn_global_load_lds(
        (const __attribute__((address_space(1))) unsigned int*)(unsigned long long)(uintptr_t)g,
        (__attribute__((address_space(3))) unsigned int*)(unsigned)(uintptr_t)lds,
        16, 0, 0);
#else
    *(u32x4*)lds = *(const u32x4*)g;
#endif
}

__device__ __forceinline__ void async_wait_all() {
#if ASYNC_MODE == 2
#if __has_builtin(__builtin_amdgcn_s_wait_asynccnt)
    __builtin_amdgcn_s_wait_asynccnt(0);
#else
    asm volatile("s_wait_asynccnt 0x0" ::: "memory");
#endif
#endif
}

// ---------------- kernel 1: fp32 -> bf16 ----------------
__global__ void cvt_f32_bf16(const float* __restrict__ src,
                             bf16_t* __restrict__ dst, size_t n) {
    size_t i = (size_t)blockIdx.x * blockDim.x + threadIdx.x;
    if (i < n) dst[i] = f2bf(src[i]);
}

// ---------------- kernel 2/5: tiled WMMA GEMM ----------------
// C[M,N] = A[M,K]*B[K,N] + bias[N].  M%64==0, N%64==0, K%32==0.
// block = 128 threads (4 waves); block tile 64x64; wave tile 16x64.
template <bool OUT_BF16>
__global__ __launch_bounds__(128)
void gemm_bf16(const bf16_t* __restrict__ A, const bf16_t* __restrict__ B,
               const float* __restrict__ bias, void* __restrict__ outv,
               int M, int Nn, int Kk) {
    __shared__ bf16_t a_t[64 * 40];   // [row 0..63][k 0..31] pad 8
    __shared__ bf16_t bT [64 * 40];   // [n  0..63][k 0..31] pad 8 (transposed)

    const int tid = threadIdx.x;
    const int w  = tid >> 5;
    const int l  = tid & 31;
    const int lh = l >> 4;            // lane half (0/1)
    const int ll = l & 15;
    const int m0 = blockIdx.y * 64;
    const int n0 = blockIdx.x * 64;

    v8f acc[4];
    #pragma unroll
    for (int t = 0; t < 4; ++t)
        #pragma unroll
        for (int r = 0; r < 8; ++r) acc[t][r] = 0.0f;

    const int arow  = tid >> 1, ahalf = tid & 1;   // A: 64 rows x 2 chunks
    const int bk    = tid >> 2, bq    = tid & 3;   // B: 32 k-rows x 4 chunks
    const int ktiles = Kk >> 5;

    for (int kt = 0; kt < ktiles; ++kt) {
        const int k0 = kt << 5;
        // --- stage A tile (row-major): async global->LDS, bypasses VGPRs ---
        const bf16_t* asrc = A + (size_t)(m0 + arow) * Kk + k0 + ahalf * 16;
        bf16_t* adst = a_t + arow * 40 + ahalf * 16;
        cp_async16(adst, asrc);
        cp_async16(adst + 8, asrc + 8);
        // --- stage B tile transposed: contiguous global read, strided LDS write ---
        V16 btmp;
        const u32x4* bs = (const u32x4*)(B + (size_t)(k0 + bk) * Nn + n0 + bq * 16);
        btmp.q[0] = bs[0]; btmp.q[1] = bs[1];
        #pragma unroll
        for (int x = 0; x < 16; ++x) bT[(bq * 16 + x) * 40 + bk] = btmp.b[x];

        if (kt + 1 < ktiles) {  // CDNA5 global_prefetch_b8 of next tiles
            __builtin_prefetch(A + (size_t)(m0 + arow) * Kk + k0 + 32, 0, 1);
            __builtin_prefetch(B + (size_t)(k0 + 32 + bk) * Nn + n0 + bq * 16, 0, 1);
        }
        async_wait_all();
        __syncthreads();

        // A fragment: 16-bit 16x32 layout (two contiguous 16B chunks per lane)
        V16 af;
        af.q[0] = *(const u32x4*)(a_t + (w * 16 + ll) * 40 + lh * 8);
        af.q[1] = *(const u32x4*)(a_t + (w * 16 + ll) * 40 + 16 + lh * 8);
        #pragma unroll
        for (int nt = 0; nt < 4; ++nt) {
            V16 bfv;  // B fragment: lane = column, two 8-k chunks (k = 8*lh.., 16+8*lh..)
            const bf16_t* br = bT + (nt * 16 + ll) * 40 + lh * 8;
            bfv.q[0] = *(const u32x4*)br; bfv.q[1] = *(const u32x4*)(br + 16);
            acc[nt] = __builtin_amdgcn_wmma_f32_16x16x32_bf16(
                false, af.v, false, bfv.v, (short)0, acc[nt], false, false);
        }
        __syncthreads();
    }

    // epilogue: C layout -> memory (row = r + 8*lh, col = lane&15)
    #pragma unroll
    for (int nt = 0; nt < 4; ++nt) {
        #pragma unroll
        for (int r = 0; r < 8; ++r) {
            const int row = m0 + w * 16 + r + 8 * lh;
            const int col = n0 + nt * 16 + ll;
            float v = acc[nt][r] + bias[col];
            if constexpr (OUT_BF16)
                ((bf16_t*)outv)[(size_t)row * Nn + col] = f2bf(v);
            else
                ((float*)outv)[(size_t)row * Nn + col] = v;
        }
    }
}

// ---------------- kernel 3: gather + 3-axis RoPE ----------------
// one thread per (serialized point i, head h, rotation pair (axis a, freq j))
__global__ void gather_rope(const bf16_t* __restrict__ qkvb,
                            const long long* __restrict__ order,
                            const int* __restrict__ grid_coord,
                            bf16_t* __restrict__ qb, bf16_t* __restrict__ kb,
                            bf16_t* __restrict__ vb) {
    size_t tid = (size_t)blockIdx.x * blockDim.x + threadIdx.x;
    if (tid >= (size_t)N_PTS * NH * 48) return;
    int pr = (int)(tid % 48);
    size_t t2 = tid / 48;
    int h = (int)(t2 % NH);
    size_t i = t2 / NH;
    int a = pr >> 4, j = pr & 15;

    long long src = order[i];
    float posa = (float)grid_coord[src * 3 + a];
    // inv_freq = 100^(-j/16) = exp(-j * ln(100)/16)
    float ang = posa * __expf(-(float)j * (4.6051701859880914f / 16.0f));
    float sv, cv;
    __sincosf(ang, &sv, &cv);

    int d = a * 32 + 2 * j;
    size_t qoff = (size_t)src * (3 * CCH) + h * DH + d;
    float q1 = (float)qkvb[qoff],            q2 = (float)qkvb[qoff + 1];
    float k1 = (float)qkvb[qoff + CCH],      k2 = (float)qkvb[qoff + CCH + 1];
    float v1 = (float)qkvb[qoff + 2 * CCH],  v2 = (float)qkvb[qoff + 2 * CCH + 1];

    size_t o = i * CCH + h * DH + d;
    qb[o]     = f2bf(q1 * cv - q2 * sv);
    qb[o + 1] = f2bf(q1 * sv + q2 * cv);
    kb[o]     = f2bf(k1 * cv - k2 * sv);
    kb[o + 1] = f2bf(k1 * sv + k2 * cv);
    vb[o]     = f2bf(v1);
    vb[o + 1] = f2bf(v2);
}

// ---------------- kernel 4: patch attention (flash-style) ----------------
#define KL_W 104                 // 96 + 8 pad (bf16 elems)
#define VT_W 264                 // 256 + 8 pad
#define KLDS_BYTES (PK * KL_W * 2)          // 53248
#define VT_BYTES   (DH * VT_W * 2)          // 50688
#define P_BYTES    (8 * 16 * 32 * 2)        // 8192
#define ATTN_SMEM  (KLDS_BYTES + VT_BYTES + P_BYTES)   // 112128 B -> 2 blocks/WGP

__global__ __launch_bounds__(256)
void attn_kernel(const bf16_t* __restrict__ qb, const bf16_t* __restrict__ kb,
                 const bf16_t* __restrict__ vb,
                 const long long* __restrict__ order,
                 bf16_t* __restrict__ ob) {
    extern __shared__ char smem[];
    bf16_t* k_lds = (bf16_t*)smem;                         // [PK][KL_W] row-major
    bf16_t* v_t   = (bf16_t*)(smem + KLDS_BYTES);          // [DH][VT_W] transposed
    bf16_t* p_lds = (bf16_t*)(smem + KLDS_BYTES + VT_BYTES); // [8 waves][16][32]

    const int blk = blockIdx.x;
    const int hf  = blk & 1;           // which 128-row half of the patch
    const int ph  = blk >> 1;
    const int h   = ph % NH;
    const int p   = ph / NH;
    const int tid = threadIdx.x;
    const int w   = tid >> 5, l = tid & 31, lh = l >> 4, ll = l & 15;

    // ---- cooperative stage: K rows async global->LDS; V manual transpose ----
    {
        const bf16_t* ks = kb + ((size_t)(p * PK + tid)) * CCH + h * DH;
        bf16_t* kd = k_lds + tid * KL_W;
        #pragma unroll
        for (int i = 0; i < 12; ++i) cp_async16(kd + i * 8, ks + i * 8);

        const bf16_t* vs = vb + ((size_t)(p * PK + tid)) * CCH + h * DH;
        bf16_t tmp[DH] __attribute__((aligned(16)));
        const u32x4* v4 = (const u32x4*)vs;
        u32x4* t4 = (u32x4*)tmp;
        #pragma unroll
        for (int i = 0; i < 12; ++i) t4[i] = v4[i];
        #pragma unroll
        for (int d = 0; d < DH; ++d) v_t[d * VT_W + tid] = tmp[d];
    }
    async_wait_all();
    __syncthreads();

    // ---- per-wave Q fragments: 16 rows x 96 (3 k-chunks of 32) ----
    const int rowbase = hf * 128 + w * 16;
    const bf16_t* qrow = qb + ((size_t)(p * PK + rowbase + ll)) * CCH + h * DH;
    v16bf qa[3];
    #pragma unroll
    for (int kk = 0; kk < 3; ++kk) {
        V16 t;
        t.q[0] = *(const u32x4*)(qrow + kk * 32 + lh * 8);
        t.q[1] = *(const u32x4*)(qrow + kk * 32 + 16 + lh * 8);
        qa[kk] = t.v;
    }

    v8f o_acc[6];
    #pragma unroll
    for (int t = 0; t < 6; ++t)
        #pragma unroll
        for (int r = 0; r < 8; ++r) o_acc[t][r] = 0.0f;
    float mrun[8], lrun[8];
    #pragma unroll
    for (int r = 0; r < 8; ++r) { mrun[r] = -1e30f; lrun[r] = 0.0f; }

    bf16_t* pw = p_lds + w * 16 * 32;   // this wave's P staging tile

    for (int j0 = 0; j0 < PK; j0 += 32) {
        // ---- scores: S[16 x 32] = Q(16x96) * K^T(96x32), two 16-col WMMAs ----
        v8f sfr[2];
        #pragma unroll
        for (int nt = 0; nt < 2; ++nt) {
            v8f s;
            #pragma unroll
            for (int r = 0; r < 8; ++r) s[r] = 0.0f;
            const int kvcol = j0 + nt * 16 + ll;   // lane's key column
            #pragma unroll
            for (int kk = 0; kk < 3; ++kk) {
                V16 bfv;   // B[k][n] = K[kvcol][kk*32 + k]: contiguous along d
                const bf16_t* br =
                    k_lds + kvcol * KL_W + kk * 32 + lh * 8;
                bfv.q[0] = *(const u32x4*)br; bfv.q[1] = *(const u32x4*)(br + 16);
                s = __builtin_amdgcn_wmma_f32_16x16x32_bf16(
                    false, qa[kk], false, bfv.v, (short)0, s, false, false);
            }
            sfr[nt] = s;
        }

        // ---- online softmax (rows live at vgpr r, lanes of half lh) ----
        float tmax[8];
        #pragma unroll
        for (int r = 0; r < 8; ++r)
            tmax[r] = fmaxf(sfr[0][r], sfr[1][r]) * SCALE_ATTN;
        #pragma unroll
        for (int m = 1; m <= 8; m <<= 1)
            #pragma unroll
            for (int r = 0; r < 8; ++r)
                tmax[r] = fmaxf(tmax[r], __shfl_xor(tmax[r], m, 32));

        float p0[8], p1[8], rs[8];
        #pragma unroll
        for (int r = 0; r < 8; ++r) {
            float mn = fmaxf(mrun[r], tmax[r]);
            float sc = __expf(mrun[r] - mn);
            p0[r] = __expf(sfr[0][r] * SCALE_ATTN - mn);
            p1[r] = __expf(sfr[1][r] * SCALE_ATTN - mn);
            rs[r] = p0[r] + p1[r];
            #pragma unroll
            for (int t = 0; t < 6; ++t) o_acc[t][r] *= sc;
            lrun[r] *= sc;
            mrun[r] = mn;
        }
        #pragma unroll
        for (int m = 1; m <= 8; m <<= 1)
            #pragma unroll
            for (int r = 0; r < 8; ++r) rs[r] += __shfl_xor(rs[r], m, 32);
        #pragma unroll
        for (int r = 0; r < 8; ++r) lrun[r] += rs[r];

        // ---- re-layout P: C-layout -> A-layout via per-wave LDS tile ----
        #pragma unroll
        for (int r = 0; r < 8; ++r) {
            const int mrow = r + 8 * lh;
            pw[mrow * 32 + ll]      = f2bf(p0[r]);
            pw[mrow * 32 + 16 + ll] = f2bf(p1[r]);
        }
        __syncthreads();
        V16 pa;
        pa.q[0] = *(const u32x4*)(pw + ll * 32 + lh * 8);
        pa.q[1] = *(const u32x4*)(pw + ll * 32 + 16 + lh * 8);

        // ---- O += P(16x32) * V(32x96): 6 column tiles, contiguous B reads ----
        #pragma unroll
        for (int t = 0; t < 6; ++t) {
            V16 vf;  // B[k][n] = V[j0+k][t*16+n] = v_t[(t*16+n)][j0+k]
            const bf16_t* vr =
                v_t + (t * 16 + ll) * VT_W + j0 + lh * 8;
            vf.q[0] = *(const u32x4*)vr; vf.q[1] = *(const u32x4*)(vr + 16);
            o_acc[t] = __builtin_amdgcn_wmma_f32_16x16x32_bf16(
                false, pa.v, false, vf.v, (short)0, o_acc[t], false, false);
        }
        __syncthreads();
    }

    // ---- normalize and scatter rows back to original point order ----
    #pragma unroll
    for (int r = 0; r < 8; ++r) {
        const float inv = 1.0f / lrun[r];
        const size_t i = (size_t)p * PK + rowbase + r + 8 * lh;
        const long long dst = order[i];
        #pragma unroll
        for (int t = 0; t < 6; ++t)
            ob[(size_t)dst * CCH + h * DH + t * 16 + ll] = f2bf(o_acc[t][r] * inv);
    }
}

// ---------------- host-side launch ----------------
extern "C" void kernel_launch(void* const* d_in, const int* in_sizes, int n_in,
                              void* d_out, int out_size, void* d_ws, size_t ws_size,
                              hipStream_t stream) {
    const float*     feat       = (const float*)d_in[0];
    const int*       grid_coord = (const int*)d_in[1];
    const long long* order      = (const long long*)d_in[2];
    const float*     W_qkv      = (const float*)d_in[3];
    const float*     b_qkv      = (const float*)d_in[4];
    const float*     W_proj     = (const float*)d_in[5];
    const float*     b_proj     = (const float*)d_in[6];
    float* out = (float*)d_out;

    char* ws = (char*)d_ws;
    const size_t nFeat = (size_t)N_PTS * CCH;
    bf16_t* featb  = (bf16_t*)ws;                         ws += nFeat * 2;
    bf16_t* wqkvb  = (bf16_t*)ws;                         ws += (size_t)CCH * 3 * CCH * 2;
    bf16_t* wprojb = (bf16_t*)ws;                         ws += (size_t)CCH * CCH * 2;
    bf16_t* qkvb   = (bf16_t*)ws;                         ws += (size_t)N_PTS * 3 * CCH * 2;
    bf16_t* qb     = (bf16_t*)ws;                         ws += nFeat * 2;
    bf16_t* kb     = (bf16_t*)ws;                         ws += nFeat * 2;
    bf16_t* vb     = (bf16_t*)ws;                         ws += nFeat * 2;
    bf16_t* ob     = (bf16_t*)ws;                         ws += nFeat * 2;

    // 1) fp32 -> bf16 conversions
    cvt_f32_bf16<<<(unsigned)((nFeat + 255) / 256), 256, 0, stream>>>(feat, featb, nFeat);
    cvt_f32_bf16<<<(unsigned)(((size_t)CCH * 3 * CCH + 255) / 256), 256, 0, stream>>>(
        W_qkv, wqkvb, (size_t)CCH * 3 * CCH);
    cvt_f32_bf16<<<(unsigned)(((size_t)CCH * CCH + 255) / 256), 256, 0, stream>>>(
        W_proj, wprojb, (size_t)CCH * CCH);

    // 2) QKV GEMM: [N,C] x [C,3C] -> bf16 qkv
    gemm_bf16<true><<<dim3(3 * CCH / 64, N_PTS / 64), 128, 0, stream>>>(
        featb, wqkvb, b_qkv, (void*)qkvb, N_PTS, 3 * CCH, CCH);

    // 3) gather + RoPE
    {
        size_t nth = (size_t)N_PTS * NH * 48;
        gather_rope<<<(unsigned)((nth + 255) / 256), 256, 0, stream>>>(
            qkvb, order, grid_coord, qb, kb, vb);
    }

    // 4) patch attention (dynamic LDS > default cap -> raise attribute; not a stream op)
    (void)hipFuncSetAttribute(reinterpret_cast<const void*>(&attn_kernel),
                              hipFuncAttributeMaxDynamicSharedMemorySize, (int)ATTN_SMEM);
    attn_kernel<<<NPATCH * NH * 2, 256, ATTN_SMEM, stream>>>(qb, kb, vb, order, ob);

    // 5) output projection: [N,C] x [C,C] + b -> fp32 out
    gemm_bf16<false><<<dim3(CCH / 64, N_PTS / 64), 128, 0, stream>>>(
        ob, wprojb, b_proj, (void*)out, N_PTS, CCH, CCH);
}
